// MambaCrossBlock_56229711839694
// MI455X (gfx1250) — compile-verified, matrix operand order checked by reading, not run on hardware
//
#include <hip/hip_runtime.h>
#include <hip/hip_bf16.h>
#include <math.h>

// ---------------------------------------------------------------------------
// MambaCrossBlock for MI455X (gfx1250, wave32, WMMA).
// All heavy GEMMs use v_wmma_f32_16x16x32_bf16 (bf16 A/B, f32 accumulate).
// Sequential SSM scans (clip => non-associative) run 16384 independent
// register-resident recurrences. Everything fits in the 192MB L2.
// ---------------------------------------------------------------------------

typedef __bf16 bf16_t;
typedef __attribute__((ext_vector_type(16))) __bf16 v16bf;
typedef __attribute__((ext_vector_type(8)))  __bf16 bf16x8;
typedef __attribute__((ext_vector_type(8)))  float  v8f;

#define EPI_BNRELU  1
#define EPI_BF16    2
#define EPI_INPROJ  3
#define EPI_RESTORE 4

// ---- 16-bit WMMA fragment load (A: row-major [rows][K]; B(weights): [N][K]).
// ISA layout (16-bit A 16x32): lanes 0-15 row=lane, k = 0..7 & 16..23;
// lanes 16-31 row=lane-16, k = 8..15 & 24..31.  B mirrors with N<->lane.
__device__ inline v16bf load_frag(const bf16_t* __restrict__ base, int row0,
                                  int k0, int ld, int lane) {
  int r    = lane & 15;
  int half = lane >> 4;
  const bf16_t* p = base + (size_t)(row0 + r) * ld + k0 + half * 8;
  bf16x8 lo = *reinterpret_cast<const bf16x8*>(p);        // k chunk +0
  bf16x8 hi = *reinterpret_cast<const bf16x8*>(p + 16);   // k chunk +16
  return __builtin_shufflevector(lo, hi, 0,1,2,3,4,5,6,7,8,9,10,11,12,13,14,15);
}

// ---- epilogue: C/D f32 16x16 layout: lanes 0-15: M=v, N=lane;
//                                      lanes 16-31: M=v+8, N=lane-16.
template <int EPI>
__device__ inline void store_tile(v8f c, int mbase, int nbase, int lane, int N,
                                  float* __restrict__ outf, bf16_t* __restrict__ outh,
                                  const float* __restrict__ bn_g, const float* __restrict__ bn_b,
                                  const float* __restrict__ bn_m, const float* __restrict__ bn_v,
                                  const float* __restrict__ resid, const float* __restrict__ gate) {
  int half = lane >> 4, col = lane & 15;
  int n = nbase + col;
  float scale = 0.f, bias = 0.f;
  if (EPI == EPI_BNRELU || EPI == EPI_RESTORE) {
    scale = bn_g[n] * rsqrtf(bn_v[n] + 1e-5f);
    bias  = bn_b[n] - bn_m[n] * scale;
  }
  float gsig = 0.f;
  if (EPI == EPI_RESTORE) {
    float gv = gate[0];
    gsig = 1.f / (1.f + expf(-gv));
  }
#pragma unroll
  for (int v = 0; v < 8; ++v) {
    int m = mbase + half * 8 + v;
    float val = c[v];
    if (EPI == EPI_BNRELU) {
      outf[(size_t)m * N + n] = fmaxf(fmaf(val, scale, bias), 0.f);
    } else if (EPI == EPI_BF16) {
      outh[(size_t)m * N + n] = (bf16_t)val;
    } else if (EPI == EPI_INPROJ) {
      if (n < 1024) {
        outf[(size_t)m * 1024 + n] = val;                       // x_ssm (f32)
      } else {
        float s = val / (1.f + expf(-val));                     // silu(gate)
        outh[(size_t)m * 1024 + (n - 1024)] = (bf16_t)s;
      }
    } else { // EPI_RESTORE: scatter to NCHW with BN + gated residual
      int b = m >> 10, l = m & 1023;
      size_t idx = ((size_t)(b * 256 + n)) * 1024 + l;
      outf[idx] = resid[idx] + gsig * fmaf(val, scale, bias);
    }
  }
}

// ---- GEMM: out[m,n] = sum_k A[m,k] * W[n,k].  Block = 8 waves (256 thr),
// block tile 128x64, wave tile 32x32 (2x2 WMMA 16x16x32 accumulators).
template <int EPI>
__global__ __launch_bounds__(256) void wmma_gemm(
    const bf16_t* __restrict__ A, const bf16_t* __restrict__ Bw,
    int K, int N,
    float* __restrict__ outf, bf16_t* __restrict__ outh,
    const float* __restrict__ bn_g, const float* __restrict__ bn_b,
    const float* __restrict__ bn_m, const float* __restrict__ bn_v,
    const float* __restrict__ resid, const float* __restrict__ gate) {
  int lane = threadIdx.x & 31;
  int wave = threadIdx.x >> 5;
  int m0 = blockIdx.x * 128 + (wave & 3) * 32;
  int n0 = blockIdx.y * 64 + (wave >> 2) * 32;

  v8f c00 = {}, c01 = {}, c10 = {}, c11 = {};
  for (int k = 0; k < K; k += 32) {
    v16bf a0 = load_frag(A,  m0,      k, K, lane);
    v16bf a1 = load_frag(A,  m0 + 16, k, K, lane);
    v16bf b0 = load_frag(Bw, n0,      k, K, lane);
    v16bf b1 = load_frag(Bw, n0 + 16, k, K, lane);
    c00 = __builtin_amdgcn_wmma_f32_16x16x32_bf16(false, a0, false, b0, (short)0, c00, false, false);
    c01 = __builtin_amdgcn_wmma_f32_16x16x32_bf16(false, a0, false, b1, (short)0, c01, false, false);
    c10 = __builtin_amdgcn_wmma_f32_16x16x32_bf16(false, a1, false, b0, (short)0, c10, false, false);
    c11 = __builtin_amdgcn_wmma_f32_16x16x32_bf16(false, a1, false, b1, (short)0, c11, false, false);
  }
  store_tile<EPI>(c00, m0,      n0,      lane, N, outf, outh, bn_g, bn_b, bn_m, bn_v, resid, gate);
  store_tile<EPI>(c01, m0,      n0 + 16, lane, N, outf, outh, bn_g, bn_b, bn_m, bn_v, resid, gate);
  store_tile<EPI>(c10, m0 + 16, n0,      lane, N, outf, outh, bn_g, bn_b, bn_m, bn_v, resid, gate);
  store_tile<EPI>(c11, m0 + 16, n0 + 16, lane, N, outf, outh, bn_g, bn_b, bn_m, bn_v, resid, gate);
}

// ---- f32 -> bf16 weight conversion -----------------------------------------
__global__ void cvt_kernel(const float* __restrict__ s, bf16_t* __restrict__ d, int n) {
  int t = blockIdx.x * 256 + threadIdx.x;
  if (t < n) d[t] = (bf16_t)s[t];
}

// ---- NCHW f32 -> token-major bf16 (LDS-tiled transpose) --------------------
__global__ __launch_bounds__(256) void nchw_to_tok_kernel(const float* __restrict__ x,
                                                          bf16_t* __restrict__ xT) {
  __shared__ float tile[32][33];
  int b  = blockIdx.z;
  int l0 = blockIdx.x * 32, c0 = blockIdx.y * 32;
  int tx = threadIdx.x & 31, ty = threadIdx.x >> 5;
#pragma unroll
  for (int jj = 0; jj < 4; ++jj) {
    int c = c0 + ty + jj * 8;
    tile[ty + jj * 8][tx] = x[((size_t)b * 256 + c) * 1024 + l0 + tx];
  }
  __syncthreads();
#pragma unroll
  for (int jj = 0; jj < 4; ++jj) {
    int l = l0 + ty + jj * 8;
    xT[((size_t)b * 1024 + l) * 256 + c0 + tx] = (bf16_t)tile[tx][ty + jj * 8];
  }
}

// ---- global average pool over spatial, conc = [V(512) | I(512)] ------------
__global__ void pool_kernel(const float* __restrict__ xrV, const float* __restrict__ xrI,
                            float* __restrict__ pooled) {
  int t = blockIdx.x * 256 + threadIdx.x;     // 4096 = B * 1024
  int b = t >> 10, c = t & 1023;
  const float* src = (c < 512) ? (xrV + ((size_t)b * 1024) * 512 + c)
                               : (xrI + ((size_t)b * 1024) * 512 + (c - 512));
  float s = 0.f;
  for (int l = 0; l < 1024; ++l) s += src[(size_t)l * 512];
  pooled[t] = s * (1.f / 1024.f);
}

// ---- lambda predictor: 1024 -> 128 (relu) -> 2 (softmax), *0.5 folded ------
__global__ __launch_bounds__(128) void lambda_kernel(
    const float* __restrict__ pooled, const float* __restrict__ w1, const float* __restrict__ b1,
    const float* __restrict__ w2, const float* __restrict__ b2, float* __restrict__ lamw) {
  __shared__ float sh[128];
  __shared__ float logits[2];
  int b = blockIdx.x, j = threadIdx.x;
  const float* p = pooled + b * 1024;
  const float* wr = w1 + j * 1024;
  float acc = b1[j];
  for (int k = 0; k < 1024; ++k) acc = fmaf(p[k], wr[k], acc);
  sh[j] = fmaxf(acc, 0.f);
  __syncthreads();
  if (j < 2) {
    float a = b2[j];
    const float* w = w2 + j * 128;
    for (int k = 0; k < 128; ++k) a = fmaf(sh[k], w[k], a);
    logits[j] = a;
  }
  __syncthreads();
  if (j == 0) {
    float mx = fmaxf(logits[0], logits[1]);
    float e0 = expf(logits[0] - mx), e1 = expf(logits[1] - mx);
    float inv = 1.f / (e0 + e1);
    lamw[b * 2 + 0] = 0.5f * e0 * inv;
    lamw[b * 2 + 1] = 0.5f * e1 * inv;
  }
}

// ---- LayerNorm over D=512, one wave32 per token, write bf16 ----------------
__global__ __launch_bounds__(256) void ln_kernel(const float* __restrict__ xr,
                                                 const float* __restrict__ g,
                                                 const float* __restrict__ b,
                                                 bf16_t* __restrict__ xn) {
  int wave = threadIdx.x >> 5, lane = threadIdx.x & 31;
  int t = blockIdx.x * 8 + wave;              // token < 4096
  const float* row = xr + (size_t)t * 512;
  float vals[16], s = 0.f, s2 = 0.f;
#pragma unroll
  for (int j = 0; j < 16; ++j) {
    float v = row[lane + j * 32];
    vals[j] = v; s += v; s2 += v * v;
  }
#pragma unroll
  for (int off = 16; off; off >>= 1) {
    s  += __shfl_xor(s,  off, 32);
    s2 += __shfl_xor(s2, off, 32);
  }
  float mu   = s * (1.f / 512.f);
  float var  = s2 * (1.f / 512.f) - mu * mu;
  float rstd = rsqrtf(var + 1e-5f);
  bf16_t* orow = xn + (size_t)t * 512;
#pragma unroll
  for (int j = 0; j < 16; ++j) {
    int cidx = lane + j * 32;
    orow[cidx] = (bf16_t)(fmaf((vals[j] - mu) * rstd, g[cidx], b[cidx]));
  }
}

// ---- SSM scan: each thread = one (stream, dir, b, i); does BOTH std and
// cross recurrences (states in registers), writes lambda-weighted sum.
__global__ __launch_bounds__(256) void scan_kernel(
    const float* __restrict__ xssm,                       // [2][4096][1024]
    const float* __restrict__ AlogV, const float* __restrict__ AlogI,
    const float* __restrict__ BV, const float* __restrict__ BI,
    const float* __restrict__ CV, const float* __restrict__ CI,
    const float* __restrict__ lamw,
    float* __restrict__ yf, float* __restrict__ yb) {
  int tid = blockIdx.x * 256 + threadIdx.x;               // 16384 total
  int i   = tid & 1023;
  int b   = (tid >> 10) & 3;
  int dir = (tid >> 12) & 1;
  int s   = (tid >> 13) & 1;
  const float* Alog = s ? AlogI : AlogV;
  const float* Bstd = s ? BI : BV;
  const float* Bcr  = s ? BV : BI;
  const float* Cm   = s ? CI : CV;

  float Aa[16], Bs[16], Bc[16], Cc[16], h1[16], h2[16];
#pragma unroll
  for (int st = 0; st < 16; ++st) {
    Aa[st] = fminf(fmaxf(-expf(Alog[i * 16 + st]), -10.f), -0.01f);
    Bs[st] = Bstd[i * 16 + st];
    Bc[st] = Bcr[i * 16 + st];
    Cc[st] = Cm[i * 16 + st];
    h1[st] = 0.f; h2[st] = 0.f;
  }
  float w0 = lamw[b * 2], w1 = lamw[b * 2 + 1];
  const float* x = xssm + ((size_t)s * 4096 + b * 1024) * 1024 + i;
  float* y = (dir ? yb : yf) + ((size_t)s * 4096 + b * 1024) * 1024 + i;

  int l = dir ? 1023 : 0, step = dir ? -1 : 1;
  for (int it = 0; it < 1024; ++it, l += step) {
    float xv = x[(size_t)l * 1024];
    float y1 = 0.f, y2 = 0.f;
#pragma unroll
    for (int st = 0; st < 16; ++st) {
      h1[st] = fminf(fmaxf(fmaf(h1[st], Aa[st], xv * Bs[st]), -10.f), 10.f);
      h2[st] = fminf(fmaxf(fmaf(h2[st], Aa[st], xv * Bc[st]), -10.f), 10.f);
      y1 = fmaf(h1[st], Cc[st], y1);
      y2 = fmaf(h2[st], Cc[st], y2);
    }
    y[(size_t)l * 1024] = fmaf(w0, y1, w1 * y2);
  }
}

// ---- combine: y16 = (yf + yb) * silu(gate)  (silu precomputed in bf16) -----
__global__ void combine_kernel(const float* __restrict__ yf, const float* __restrict__ yb,
                               const bf16_t* __restrict__ gs, bf16_t* __restrict__ y16) {
  size_t t = (size_t)blockIdx.x * 256 + threadIdx.x;      // 2*4096*1024
  y16[t] = (bf16_t)((yf[t] + yb[t]) * (float)gs[t]);
}

// ---------------------------------------------------------------------------
extern "C" void kernel_launch(void* const* d_in, const int* in_sizes, int n_in,
                              void* d_out, int out_size, void* d_ws, size_t ws_size,
                              hipStream_t stream) {
  (void)in_sizes; (void)n_in; (void)out_size; (void)ws_size;
  const float* x_V       = (const float*)d_in[0];
  const float* x_I       = (const float*)d_in[1];
  const float* conv_red_w= (const float*)d_in[2];
  const float* bn_red_g  = (const float*)d_in[3];
  const float* bn_red_b  = (const float*)d_in[4];
  const float* bn_red_m  = (const float*)d_in[5];
  const float* bn_red_v  = (const float*)d_in[6];
  const float* conv_res_w= (const float*)d_in[7];
  const float* bn_res_g  = (const float*)d_in[8];
  const float* bn_res_b  = (const float*)d_in[9];
  const float* bn_res_m  = (const float*)d_in[10];
  const float* bn_res_v  = (const float*)d_in[11];
  const float* lam_w1    = (const float*)d_in[12];
  const float* lam_b1    = (const float*)d_in[13];
  const float* lam_w2    = (const float*)d_in[14];
  const float* lam_b2    = (const float*)d_in[15];
  const float* V_in_w    = (const float*)d_in[16];
  const float* V_out_w   = (const float*)d_in[17];
  const float* V_A_log   = (const float*)d_in[18];
  const float* V_B       = (const float*)d_in[19];
  const float* V_C       = (const float*)d_in[20];
  const float* V_ln_g    = (const float*)d_in[21];
  const float* V_ln_b    = (const float*)d_in[22];
  const float* I_in_w    = (const float*)d_in[23];
  const float* I_out_w   = (const float*)d_in[24];
  const float* I_A_log   = (const float*)d_in[25];
  const float* I_B       = (const float*)d_in[26];
  const float* I_C       = (const float*)d_in[27];
  const float* I_ln_g    = (const float*)d_in[28];
  const float* I_ln_b    = (const float*)d_in[29];
  const float* gate      = (const float*)d_in[30];
  float* out = (float*)d_out;

  // ---- workspace arena ----
  char* w = (char*)d_ws;
  size_t off = 0;
  auto take = [&](size_t bytes) -> void* {
    void* p = w + off;
    off += (bytes + 255) & ~(size_t)255;
    return p;
  };
  const size_t TOK = 4096;                    // B * L
  bf16_t* wRed  = (bf16_t*)take(512 * 256 * 2);
  bf16_t* wRes  = (bf16_t*)take(256 * 512 * 2);
  bf16_t* wIn   = (bf16_t*)take(2 * 2048 * 512 * 2);
  bf16_t* wOut  = (bf16_t*)take(2 * 512 * 1024 * 2);
  bf16_t* xT    = (bf16_t*)take(2 * TOK * 256 * 2);
  float*  xr    = (float*) take(2 * TOK * 512 * 4);
  bf16_t* xn    = (bf16_t*)take(2 * TOK * 512 * 2);
  float*  xssm  = (float*) take(2 * TOK * 1024 * 4);
  bf16_t* gsilu = (bf16_t*)take(2 * TOK * 1024 * 2);
  float*  yfb   = (float*) take(2 * TOK * 1024 * 4);
  float*  ybb   = (float*) take(2 * TOK * 1024 * 4);
  bf16_t* y16   = (bf16_t*)take(2 * TOK * 1024 * 2);
  bf16_t* feat  = (bf16_t*)take(2 * TOK * 512 * 2);
  float*  pooled= (float*) take(4 * 1024 * 4);
  float*  lamw  = (float*) take(4 * 2 * 4);

  // ---- 1. weights -> bf16 ----
  auto cvt = [&](const float* s, bf16_t* d, int n) {
    cvt_kernel<<<(n + 255) / 256, 256, 0, stream>>>(s, d, n);
  };
  cvt(conv_red_w, wRed, 512 * 256);
  cvt(conv_res_w, wRes, 256 * 512);
  cvt(V_in_w,  wIn,               2048 * 512);
  cvt(I_in_w,  wIn + 2048 * 512,  2048 * 512);
  cvt(V_out_w, wOut,              512 * 1024);
  cvt(I_out_w, wOut + 512 * 1024, 512 * 1024);

  // ---- 2. NCHW -> token-major bf16 ----
  nchw_to_tok_kernel<<<dim3(32, 8, 4), 256, 0, stream>>>(x_V, xT);
  nchw_to_tok_kernel<<<dim3(32, 8, 4), 256, 0, stream>>>(x_I, xT + TOK * 256);

  // ---- 3. channel reduce GEMM + BN + ReLU -> xr (f32) ----
  for (int s = 0; s < 2; ++s)
    wmma_gemm<EPI_BNRELU><<<dim3(32, 8), 256, 0, stream>>>(
        xT + (size_t)s * TOK * 256, wRed, 256, 512,
        xr + (size_t)s * TOK * 512, nullptr,
        bn_red_g, bn_red_b, bn_red_m, bn_red_v, nullptr, nullptr);

  // ---- 4. pool + lambda predictor (lamV == lamI; 0.5 factor folded) ----
  pool_kernel<<<16, 256, 0, stream>>>(xr, xr + TOK * 512, pooled);
  lambda_kernel<<<4, 128, 0, stream>>>(pooled, lam_w1, lam_b1, lam_w2, lam_b2, lamw);

  // ---- 5. LayerNorm -> xn (bf16) ----
  ln_kernel<<<512, 256, 0, stream>>>(xr,             V_ln_g, V_ln_b, xn);
  ln_kernel<<<512, 256, 0, stream>>>(xr + TOK * 512, I_ln_g, I_ln_b, xn + TOK * 512);

  // ---- 6. in-projection GEMM -> x_ssm (f32) + silu(gate) (bf16) ----
  for (int s = 0; s < 2; ++s)
    wmma_gemm<EPI_INPROJ><<<dim3(32, 32), 256, 0, stream>>>(
        xn + (size_t)s * TOK * 512, wIn + (size_t)s * 2048 * 512, 512, 2048,
        xssm + (size_t)s * TOK * 1024, gsilu + (size_t)s * TOK * 1024,
        nullptr, nullptr, nullptr, nullptr, nullptr, nullptr);

  // ---- 7. SSM scans (fwd+bwd, std+cross fused per thread) ----
  scan_kernel<<<64, 256, 0, stream>>>(xssm, V_A_log, I_A_log, V_B, I_B, V_C, I_C,
                                      lamw, yfb, ybb);

  // ---- 8. combine -> y16 (bf16) ----
  combine_kernel<<<32768, 256, 0, stream>>>(yfb, ybb, gsilu, y16);

  // ---- 9. out-projection GEMM -> feat (bf16) ----
  for (int s = 0; s < 2; ++s)
    wmma_gemm<EPI_BF16><<<dim3(32, 8), 256, 0, stream>>>(
        y16 + (size_t)s * TOK * 1024, wOut + (size_t)s * 512 * 1024, 1024, 512,
        nullptr, feat + (size_t)s * TOK * 512,
        nullptr, nullptr, nullptr, nullptr, nullptr, nullptr);

  // ---- 10. channel restore GEMM + BN + gated residual -> d_out (NCHW) ----
  wmma_gemm<EPI_RESTORE><<<dim3(32, 4), 256, 0, stream>>>(
      feat, wRes, 512, 256, out, nullptr,
      bn_res_g, bn_res_b, bn_res_m, bn_res_v, x_V, gate);
  wmma_gemm<EPI_RESTORE><<<dim3(32, 4), 256, 0, stream>>>(
      feat + TOK * 512, wRes, 512, 256, out + (size_t)4 * 256 * 1024, nullptr,
      bn_res_g, bn_res_b, bn_res_m, bn_res_v, x_I, gate);
}
